// MixtralMoE_55070070669327
// MI455X (gfx1250) — matrix operand order repaired and checked by reading, not compiled, not run on hardware
//
#include <hip/hip_runtime.h>
#include <hip/hip_bf16.h>

// ---------------------------------------------------------------------------
// CDNA5 (gfx1250) sparse top-2 MoE forward with bf16 WMMA (f32 accumulate),
// async global->LDS staging for the bf16 activation tile in the down-proj.
// ---------------------------------------------------------------------------

typedef __attribute__((ext_vector_type(16))) __bf16 v16bf;
typedef __attribute__((ext_vector_type(8)))  float  v8f;

union FragAB { v16bf v; uint4 u4[2]; };
union FragC  { v8f   v; float f[8]; };

__device__ __forceinline__ unsigned short f2bf(float f) {
  // round-to-nearest-even fp32 -> bf16
  unsigned int u = __float_as_uint(f);
  unsigned int r = u + 0x7FFFu + ((u >> 16) & 1u);
  return (unsigned short)(r >> 16);
}

__device__ __forceinline__ v8f wmma_bf16(v16bf a, v16bf b, v8f c) {
  // (neg_a, A, neg_b, B, c_mod, C, reuse_a, reuse_b)
  return __builtin_amdgcn_wmma_f32_16x16x32_bf16(false, a, false, b, (short)0, c,
                                                 false, false);
}

// LDS byte offset of a __shared__ object: for the LDS aperture, addr[31:0] is
// the wave-relative LDS offset (ISA 10.2 aperture table).
__device__ __forceinline__ unsigned lds_off_u32(const void* p) {
  return (unsigned)(unsigned long long)p;
}

// Async 16B global -> LDS copy per lane (tracked on ASYNCcnt, no VGPR data).
__device__ __forceinline__ void async_copy_b128(unsigned lds_off, const void* gaddr) {
  asm volatile("global_load_async_to_lds_b128 %0, %1, off"
               :: "v"(lds_off), "v"(gaddr) : "memory");
}

__device__ __forceinline__ void wait_asynccnt0() {
  asm volatile("s_wait_asynccnt 0x0" ::: "memory");
}

// ---------------------------------------------------------------------------
// 1) Router: one wave32 per token. logits = x @ gate_w^T, softmax, top-2.
// ---------------------------------------------------------------------------
__global__ __launch_bounds__(256) void moe_router(
    const float* __restrict__ x, const float* __restrict__ gw,
    int* __restrict__ eidx, float* __restrict__ ewgt, int T, int H)
{
  const int lane = threadIdx.x & 31;
  const int wid  = threadIdx.x >> 5;
  const int t = blockIdx.x * 8 + wid;
  if (t >= T) return;

  float acc[8];
  #pragma unroll
  for (int e = 0; e < 8; ++e) acc[e] = 0.f;

  for (int h = lane; h < H; h += 32) {
    const float xv = x[(size_t)t * H + h];
    #pragma unroll
    for (int e = 0; e < 8; ++e) acc[e] += xv * gw[(size_t)e * H + h];
  }
  #pragma unroll
  for (int off = 16; off > 0; off >>= 1) {
    #pragma unroll
    for (int e = 0; e < 8; ++e) acc[e] += __shfl_xor(acc[e], off, 32);
  }
  if (lane == 0) {
    float mx = acc[0];
    #pragma unroll
    for (int e = 1; e < 8; ++e) mx = fmaxf(mx, acc[e]);
    float p[8], s = 0.f;
    #pragma unroll
    for (int e = 0; e < 8; ++e) { p[e] = __expf(acc[e] - mx); s += p[e]; }
    const float inv_s = 1.f / s;
    #pragma unroll
    for (int e = 0; e < 8; ++e) p[e] *= inv_s;
    int a0 = 0;
    #pragma unroll
    for (int e = 1; e < 8; ++e) if (p[e] > p[a0]) a0 = e;
    int a1 = (a0 == 0) ? 1 : 0;
    #pragma unroll
    for (int e = 0; e < 8; ++e) if (e != a0 && p[e] > p[a1]) a1 = e;
    const float inv = 1.f / (p[a0] + p[a1]);
    eidx[2 * t + 0] = a0;  ewgt[2 * t + 0] = p[a0] * inv;
    eidx[2 * t + 1] = a1;  ewgt[2 * t + 1] = p[a1] * inv;
  }
}

// ---------------------------------------------------------------------------
// 2) Build compact per-expert token lists.
// ---------------------------------------------------------------------------
__global__ void moe_count(const int* __restrict__ eidx, int* __restrict__ counts,
                          int T) {
  const int t = blockIdx.x * blockDim.x + threadIdx.x;
  if (t < T) {
    atomicAdd(&counts[eidx[2 * t + 0]], 1);
    atomicAdd(&counts[eidx[2 * t + 1]], 1);
  }
}

__global__ void moe_scan(const int* __restrict__ counts, int* __restrict__ base,
                         int E) {
  if (threadIdx.x == 0 && blockIdx.x == 0) {
    int s = 0;
    for (int e = 0; e < E; ++e) { base[e] = s; s += counts[e]; }
  }
}

__global__ void moe_place(const int* __restrict__ eidx, const float* __restrict__ ewgt,
                          const int* __restrict__ base, int* __restrict__ cnt2,
                          int* __restrict__ list_slot, float* __restrict__ list_w,
                          int T) {
  const int t = blockIdx.x * blockDim.x + threadIdx.x;
  if (t < T) {
    #pragma unroll
    for (int k = 0; k < 2; ++k) {
      const int e = eidx[2 * t + k];
      const int p = atomicAdd(&cnt2[e], 1);
      const int idx = base[e] + p;
      list_slot[idx] = 2 * t + k;      // unique (token, k) output slot
      list_w[idx]    = ewgt[2 * t + k];
    }
  }
}

// ---------------------------------------------------------------------------
// 3) Up projection: g = silu(X@W1^T) * (X@W3^T) * route_w, bf16 output.
//    Block = 256 thr (8 waves, 2x4), tile 64(M) x 128(N over F), K-step 32.
// ---------------------------------------------------------------------------
__global__ __launch_bounds__(256) void moe_up(
    const float* __restrict__ x, const float* __restrict__ w1,
    const float* __restrict__ w3,
    const int* __restrict__ list_slot, const float* __restrict__ list_w,
    const int* __restrict__ counts, const int* __restrict__ base,
    unsigned short* __restrict__ g_buf, int H, int F)
{
  __shared__ unsigned short lds_x [64  * 32];
  __shared__ unsigned short lds_w1[128 * 32];
  __shared__ unsigned short lds_w3[128 * 32];
  __shared__ int   lds_tok[64];
  __shared__ float lds_wgt[64];

  const int e   = blockIdx.z;
  const int cnt = counts[e];
  const int m0  = blockIdx.x * 64;
  if (m0 >= cnt) return;                 // uniform early-exit
  const int n0  = blockIdx.y * 128;
  const int rb  = base[e];
  const int tid = threadIdx.x;
  const int lane = tid & 31, wid = tid >> 5;
  const int wm = wid & 1, wn = wid >> 1;           // 2x4 wave grid
  const int laneLo = lane & 15, laneHi = lane >> 4;

  if (tid < 64) {
    const int r  = m0 + tid;
    const int ok = (r < cnt);
    const int slot = ok ? list_slot[rb + r] : 0;
    lds_tok[tid] = slot >> 1;                      // token id
    lds_wgt[tid] = ok ? list_w[rb + r] : 0.f;
  }

  const float* w1e = w1 + (size_t)e * F * H;
  const float* w3e = w3 + (size_t)e * F * H;

  FragC c1[2][2], c3[2][2];
  #pragma unroll
  for (int mi = 0; mi < 2; ++mi)
    #pragma unroll
    for (int ni = 0; ni < 2; ++ni)
      #pragma unroll
      for (int i = 0; i < 8; ++i) { c1[mi][ni].f[i] = 0.f; c3[mi][ni].f[i] = 0.f; }

  for (int k0 = 0; k0 < H; k0 += 32) {
    __syncthreads();
    // X tile 64x32 fp32->bf16 (gathered token rows)
    #pragma unroll
    for (int it = 0; it < 2; ++it) {
      const int q = tid + it * 256;                // 512 float4 quads
      const int row = q >> 3, c4 = (q & 7) << 2;
      const float4 v = *(const float4*)(x + (size_t)lds_tok[row] * H + k0 + c4);
      unsigned short* d = &lds_x[row * 32 + c4];
      d[0] = f2bf(v.x); d[1] = f2bf(v.y); d[2] = f2bf(v.z); d[3] = f2bf(v.w);
    }
    // W1/W3 tiles 128x32 fp32->bf16
    #pragma unroll
    for (int it = 0; it < 4; ++it) {
      const int q = tid + it * 256;                // 1024 quads per tile
      const int row = q >> 3, c4 = (q & 7) << 2;
      const size_t go = (size_t)(n0 + row) * H + k0 + c4;
      { const float4 v = *(const float4*)(w1e + go);
        unsigned short* d = &lds_w1[row * 32 + c4];
        d[0] = f2bf(v.x); d[1] = f2bf(v.y); d[2] = f2bf(v.z); d[3] = f2bf(v.w); }
      { const float4 v = *(const float4*)(w3e + go);
        unsigned short* d = &lds_w3[row * 32 + c4];
        d[0] = f2bf(v.x); d[1] = f2bf(v.y); d[2] = f2bf(v.z); d[3] = f2bf(v.w); }
    }
    if (k0 + 32 < H)                               // global_prefetch_b8
      __builtin_prefetch(w1e + (size_t)(n0 + (tid >> 1)) * H + k0 + 32, 0, 1);
    __syncthreads();

    // A frags: lane<16 -> M=lane, K={0..7,16..23}; lane>=16 -> K={8..15,24..31}
    FragAB a[2], b1f[2], b3f[2];
    #pragma unroll
    for (int mi = 0; mi < 2; ++mi) {
      const int row = wm * 32 + mi * 16 + laneLo;
      const int kb  = laneHi ? 8 : 0;
      a[mi].u4[0] = *(const uint4*)&lds_x[row * 32 + kb];
      a[mi].u4[1] = *(const uint4*)&lds_x[row * 32 + kb + 16];
    }
    // B frags: lane -> N=laneLo, K={0..7,16..23} (lane<16) / {8..15,24..31}
    #pragma unroll
    for (int ni = 0; ni < 2; ++ni) {
      const int row = wn * 32 + ni * 16 + laneLo;
      const int kb  = laneHi ? 8 : 0;
      b1f[ni].u4[0] = *(const uint4*)&lds_w1[row * 32 + kb];
      b1f[ni].u4[1] = *(const uint4*)&lds_w1[row * 32 + kb + 16];
      b3f[ni].u4[0] = *(const uint4*)&lds_w3[row * 32 + kb];
      b3f[ni].u4[1] = *(const uint4*)&lds_w3[row * 32 + kb + 16];
    }
    #pragma unroll
    for (int mi = 0; mi < 2; ++mi)
      #pragma unroll
      for (int ni = 0; ni < 2; ++ni) {
        c1[mi][ni].v = wmma_bf16(a[mi].v, b1f[ni].v, c1[mi][ni].v);
        c3[mi][ni].v = wmma_bf16(a[mi].v, b3f[ni].v, c3[mi][ni].v);
      }
  }

  // Epilogue: fused SiLU * h3 * routing weight, store bf16 g.
  #pragma unroll
  for (int mi = 0; mi < 2; ++mi)
    #pragma unroll
    for (int ni = 0; ni < 2; ++ni)
      #pragma unroll
      for (int r = 0; r < 8; ++r) {
        const int rloc = wm * 32 + mi * 16 + laneHi * 8 + r;
        if (m0 + rloc < cnt) {
          const float h1v = c1[mi][ni].f[r];
          const float h3v = c3[mi][ni].f[r];
          const float g = h1v / (1.f + __expf(-h1v)) * h3v * lds_wgt[rloc];
          const int fc = n0 + wn * 32 + ni * 16 + laneLo;
          g_buf[(size_t)(rb + m0 + rloc) * F + fc] = f2bf(g);
        }
      }
}

// ---------------------------------------------------------------------------
// 4) Down projection: Y = G @ W2^T (K = F). A rows are contiguous list rows.
//    G tile is already bf16: staged with GLOBAL_LOAD_ASYNC_TO_LDS_B128
//    (ASYNCcnt) while the W2 tile is converted fp32->bf16 through VGPRs.
// ---------------------------------------------------------------------------
__global__ __launch_bounds__(256) void moe_down(
    const unsigned short* __restrict__ g_buf, const float* __restrict__ w2,
    const int* __restrict__ list_slot,
    const int* __restrict__ counts, const int* __restrict__ base,
    float* __restrict__ temp, int H, int F)
{
  __shared__ unsigned short lds_g[64  * 32];
  __shared__ unsigned short lds_w[128 * 32];
  __shared__ int lds_slot[64];

  const int e   = blockIdx.z;
  const int cnt = counts[e];
  const int m0  = blockIdx.x * 64;
  if (m0 >= cnt) return;
  const int n0  = blockIdx.y * 128;
  const int rb  = base[e];
  const int tid = threadIdx.x;
  const int lane = tid & 31, wid = tid >> 5;
  const int wm = wid & 1, wn = wid >> 1;
  const int laneLo = lane & 15, laneHi = lane >> 4;

  if (tid < 64) {
    const int r = m0 + tid;
    lds_slot[tid] = (r < cnt) ? list_slot[rb + r] : 0;
  }

  const float* w2e = w2 + (size_t)e * H * F;

  // Per-thread fixed targets for the async G-tile copy (one 16B chunk each).
  const int grow = tid >> 2, gc8 = (tid & 3) << 3;
  const unsigned g_lds = lds_off_u32(&lds_g[grow * 32 + gc8]);
  const unsigned short* g_src =
      g_buf + (size_t)(rb + m0 + grow) * F + gc8;

  FragC c[2][2];
  #pragma unroll
  for (int mi = 0; mi < 2; ++mi)
    #pragma unroll
    for (int ni = 0; ni < 2; ++ni)
      #pragma unroll
      for (int i = 0; i < 8; ++i) c[mi][ni].f[i] = 0.f;

  for (int k0 = 0; k0 < F; k0 += 32) {
    __syncthreads();
    // G tile 64x32 bf16: async global->LDS (no VGPR round-trip, ASYNCcnt)
    async_copy_b128(g_lds, g_src + k0);
    // W2 tile 128x32 fp32->bf16 through VGPRs (overlaps the async copy)
    #pragma unroll
    for (int it = 0; it < 4; ++it) {
      const int q = tid + it * 256;
      const int row = q >> 3, c4 = (q & 7) << 2;
      const float4 v = *(const float4*)(w2e + (size_t)(n0 + row) * F + k0 + c4);
      unsigned short* d = &lds_w[row * 32 + c4];
      d[0] = f2bf(v.x); d[1] = f2bf(v.y); d[2] = f2bf(v.z); d[3] = f2bf(v.w);
    }
    if (k0 + 32 < F)
      __builtin_prefetch(w2e + (size_t)(n0 + (tid >> 1)) * F + k0 + 32, 0, 1);
    wait_asynccnt0();      // this wave's async LDS writes are done
    __syncthreads();       // all waves' tiles visible

    FragAB a[2], b[2];
    #pragma unroll
    for (int mi = 0; mi < 2; ++mi) {
      const int row = wm * 32 + mi * 16 + laneLo;
      const int kb  = laneHi ? 8 : 0;
      a[mi].u4[0] = *(const uint4*)&lds_g[row * 32 + kb];
      a[mi].u4[1] = *(const uint4*)&lds_g[row * 32 + kb + 16];
    }
    #pragma unroll
    for (int ni = 0; ni < 2; ++ni) {
      const int row = wn * 32 + ni * 16 + laneLo;
      const int kb  = laneHi ? 8 : 0;
      b[ni].u4[0] = *(const uint4*)&lds_w[row * 32 + kb];
      b[ni].u4[1] = *(const uint4*)&lds_w[row * 32 + kb + 16];
    }
    #pragma unroll
    for (int mi = 0; mi < 2; ++mi)
      #pragma unroll
      for (int ni = 0; ni < 2; ++ni)
        c[mi][ni].v = wmma_bf16(a[mi].v, b[ni].v, c[mi][ni].v);
  }

  // Scatter rows to per-(token,k) slots (each slot written exactly once).
  #pragma unroll
  for (int mi = 0; mi < 2; ++mi)
    #pragma unroll
    for (int ni = 0; ni < 2; ++ni)
      #pragma unroll
      for (int r = 0; r < 8; ++r) {
        const int rloc = wm * 32 + mi * 16 + laneHi * 8 + r;
        if (m0 + rloc < cnt) {
          const int slot = lds_slot[rloc];
          const int hc = n0 + wn * 32 + ni * 16 + laneLo;
          temp[(size_t)slot * H + hc] = c[mi][ni].f[r];
        }
      }
}

// ---------------------------------------------------------------------------
// 5) Deterministic combine: out[t] = slot(2t) + slot(2t+1).
// ---------------------------------------------------------------------------
__global__ void moe_combine(const float* __restrict__ temp,
                            float* __restrict__ out, int H, int n) {
  const int i = blockIdx.x * 256 + threadIdx.x;
  if (i < n) {
    const int t = i / H, h = i - t * H;
    out[i] = temp[(size_t)(2 * t) * H + h] + temp[(size_t)(2 * t + 1) * H + h];
  }
}

// ---------------------------------------------------------------------------
extern "C" void kernel_launch(void* const* d_in, const int* in_sizes, int n_in,
                              void* d_out, int out_size, void* d_ws, size_t ws_size,
                              hipStream_t stream)
{
  const float* x  = (const float*)d_in[0];  // [B,S,H]
  const float* gw = (const float*)d_in[1];  // [E,H]
  const float* w1 = (const float*)d_in[2];  // [E,F,H]
  const float* w2 = (const float*)d_in[3];  // [E,H,F]
  const float* w3 = (const float*)d_in[4];  // [E,F,H]

  const int E = 8;
  const int H = in_sizes[1] / E;            // 1024
  const int F = in_sizes[2] / (E * H);      // 3584
  const int T = in_sizes[0] / H;            // 4096

  char* ws = (char*)d_ws;
  size_t off = 0;
  auto carve = [&](size_t bytes) -> void* {
    void* p = ws + off;
    off = (off + bytes + 255) & ~(size_t)255;
    return p;
  };
  int*   eidx      = (int*)  carve((size_t)2 * T * 4);
  float* ewgt      = (float*)carve((size_t)2 * T * 4);
  int*   counts    = (int*)  carve((size_t)2 * E * 4);   // counts | cnt2
  int*   cnt2      = counts + E;
  int*   base      = (int*)  carve((size_t)E * 4);
  int*   list_slot = (int*)  carve((size_t)2 * T * 4);
  float* list_w    = (float*)carve((size_t)2 * T * 4);
  unsigned short* g_buf = (unsigned short*)carve(((size_t)2 * T + 64) * F * 2);
  float* temp      = (float*)carve((size_t)2 * T * H * 4);
  (void)ws_size; (void)n_in; (void)out_size;

  hipMemsetAsync(counts, 0, 2 * E * sizeof(int), stream);
  moe_router <<<T / 8,            256, 0, stream>>>(x, gw, eidx, ewgt, T, H);
  moe_count  <<<(T + 255) / 256,  256, 0, stream>>>(eidx, counts, T);
  moe_scan   <<<1,                1,   0, stream>>>(counts, base, E);
  moe_place  <<<(T + 255) / 256,  256, 0, stream>>>(eidx, ewgt, base, cnt2,
                                                    list_slot, list_w, T);
  dim3 gu(T / 64, F / 128, E);
  moe_up  <<<gu, 256, 0, stream>>>(x, w1, w3, list_slot, list_w, counts, base,
                                   g_buf, H, F);
  dim3 gd(T / 64, H / 128, E);
  moe_down<<<gd, 256, 0, stream>>>(g_buf, w2, list_slot, counts, base, temp, H, F);
  moe_combine<<<(T * H + 255) / 256, 256, 0, stream>>>(temp, (float*)d_out, H, T * H);
}
